// MultiHeadSelfAttention_87574383165904
// MI455X (gfx1250) — matrix operand order repaired and checked by reading, not compiled, not run on hardware
//
#include <hip/hip_runtime.h>
#include <hip/hip_bf16.h>

typedef __attribute__((ext_vector_type(16))) __bf16 v16bf;
typedef __attribute__((ext_vector_type(8)))  float  v8f;
typedef unsigned int u32;
typedef __attribute__((ext_vector_type(4))) u32 u32x4;
typedef __attribute__((ext_vector_type(8))) int i32x8;
typedef __attribute__((ext_vector_type(4))) int i32x4;

#define B_     2
#define S_     2048
#define D_     1024
#define H_     16
#define DEPTH_ 64
#define M_     (B_ * S_)   // 4096 rows of x
#define N3_    (3 * D_)    // 3072

// log2(e) and softmax scale folded into exp2 domain
#define SCALE_LOG2E 0.1803368801f   // (1/sqrt(64)) * log2(e)
#define MASK_LOG2E  (-1.442695041e9f)

// ---------- fragment loaders ----------
// B-frag (32x16 bf16): lane holds k = kb..kb+7 and kb+16..kb+23 (two 16B chunks).
__device__ __forceinline__ v16bf load16(const __bf16* p) {
  union { uint4 u[2]; v16bf v; } c;
  c.u[0] = *(const uint4*)(p);
  c.u[1] = *(const uint4*)(p + 16);
  return c.v;
}
// A-frag (16x32 bf16): lane holds k = kb..kb+7 and kb+16..kb+23 (two 16B chunks).
__device__ __forceinline__ v16bf load8x2(const __bf16* p0, const __bf16* p1) {
  union { uint4 u[2]; v16bf v; } c;
  c.u[0] = *(const uint4*)(p0);
  c.u[1] = *(const uint4*)(p1);
  return c.v;
}

__device__ __forceinline__ v8f wmma_bf16(v16bf a, v16bf b, v8f c) {
  return __builtin_amdgcn_wmma_f32_16x16x32_bf16(false, a, false, b, (short)0, c,
                                                 false, false);
}

// gfx1250 async copy global -> LDS, tracked by ASYNCcnt.
// Low 32 bits of a generic pointer to __shared__ == LDS byte offset (aperture rule).
__device__ __forceinline__ void async_b128(const __bf16* g, __bf16* l) {
  unsigned lds = (unsigned)(size_t)l;
  asm volatile("global_load_async_to_lds_b128 %0, %1, off"
               :: "v"(lds), "v"(g) : "memory");
}
#define WAIT_ASYNC0() asm volatile("s_wait_asynccnt 0x0" ::: "memory")

// ---------- TDM: one descriptor moves a whole 2D bf16 tile to LDS ----------
// D# group0: [1:0]=count(1), [63:32]=lds_addr, [120:64]=global_addr, [127:126]=type(2)
// D# group1: [17:16]=data_size(1 => 2B), [79:48]=tensor_dim0, [111:80]=tensor_dim1,
//            [127:112]=tile_dim0, [143:128]=tile_dim1, [207:160]=tensor_dim0_stride
__device__ __forceinline__ void tdm_load_2d(unsigned lds_addr, const void* gaddr,
                                            u32 tile_d0, u32 tile_d1,
                                            u32 tensor_d0, u32 tensor_d1,
                                            u32 stride0_elems) {
  unsigned long long ga = (unsigned long long)(size_t)gaddr;
  u32x4 g0;
  g0[0] = 1u;                                             // count=1 (valid user D#)
  g0[1] = lds_addr;                                       // LDS byte address
  g0[2] = (u32)ga;                                        // global addr [31:0]
  g0[3] = (u32)((ga >> 32) & 0x01FFFFFFu) | 0x80000000u;  // addr[56:32] | type=2
  i32x8 g1;
  g1[0] = (int)(1u << 16);                                // data_size=1 (2 bytes)
  g1[1] = (int)((tensor_d0 & 0xFFFFu) << 16);             // td0[15:0]
  g1[2] = (int)((tensor_d0 >> 16) | ((tensor_d1 & 0xFFFFu) << 16));
  g1[3] = (int)((tensor_d1 >> 16) | ((tile_d0 & 0xFFFFu) << 16));
  g1[4] = (int)(tile_d1 & 0xFFFFu);                       // tile_d1, tile_d2=0
  g1[5] = (int)stride0_elems;                             // dim0 stride [31:0]
  g1[6] = 0;                                              // stride hi | dim1 stride lo
  g1[7] = 0;
  i32x4 z4 = {};
#if __clang_major__ >= 23
  i32x8 z8 = {};
  __builtin_amdgcn_tensor_load_to_lds(g0, g1, z4, z4, z8, 0);
#else
  __builtin_amdgcn_tensor_load_to_lds(g0, g1, z4, z4, 0);
#endif
}

// ---------- prep kernels ----------
__global__ __launch_bounds__(256) void k_cvt_bf16(const float* __restrict__ src,
                                                  __bf16* __restrict__ dst, int n) {
  int i = blockIdx.x * blockDim.x + threadIdx.x;
  if (i < n) dst[i] = (__bf16)src[i];
}

// dst[n*K + k] = (bf16) src[k*N + n]   (weights: [K][N] f32 -> [N][K] bf16)
__global__ __launch_bounds__(256) void k_transpose_bf16(const float* __restrict__ src,
                                                        __bf16* __restrict__ dst,
                                                        int K, int N) {
  int i = blockIdx.x * blockDim.x + threadIdx.x;
  if (i < K * N) {
    int k = i % K, n = i / K;
    dst[i] = (__bf16)src[(size_t)k * N + n];
  }
}

// ---------- QKV projection: [4096,1024] @ [1024,3072] + bias ----------
// Block = 8 waves; wave owns a 32x64 tile (two 16-row A-frags share each B-frag).
// B tile (64 cols x 32 k = 4KB) staged in LDS via async copy, double-buffered,
// single barrier per chunk (copy overlaps full compute of previous chunk).
__global__ __launch_bounds__(256) void k_gemm_qkv(
    const __bf16* __restrict__ xb, const __bf16* __restrict__ wT,
    const float* __restrict__ bias,
    __bf16* __restrict__ Qb, __bf16* __restrict__ Kb, __bf16* __restrict__ Vt) {
  __shared__ __align__(16) __bf16 ldsB[2][64 * 32];
  const int lane  = threadIdx.x & 31;
  const int warp  = threadIdx.x >> 5;
  const int NTG   = N3_ / 64;           // 48
  const int mtg   = blockIdx.x / NTG;
  const int ng    = blockIdx.x % NTG;
  const int mbase = (mtg * 8 + warp) * 32;
  const int row16 = lane & 15;
  const int hi    = lane >> 4;
  const int akb   = hi ? 8 : 0;
  const int bkb   = hi ? 8 : 0;
  const int nbase = ng * 64;

  // cooperative B staging: thread -> one 16B segment (64 rows x 32 k)
  const int sr = threadIdx.x >> 2;        // 0..63 (row n within tile)
  const int sc = (threadIdx.x & 3) * 8;   // 0,8,16,24 (k element)
  auto stageB = [&](int k0, int bufi) {
    async_b128(wT + (size_t)(nbase + sr) * D_ + k0 + sc, &ldsB[bufi][sr * 32 + sc]);
  };

  v8f acc[2][4] = {};
  const __bf16* arow0 = xb + (size_t)(mbase + row16) * D_;
  const __bf16* arow1 = arow0 + (size_t)16 * D_;

  stageB(0, 0);
  int cur = 0;
  for (int k0 = 0; k0 < D_; k0 += 32) {
    // issue A loads early: latency overlaps async wait + barrier
    v16bf a0 = load8x2(arow0 + k0 + akb, arow0 + k0 + akb + 16);
    v16bf a1 = load8x2(arow1 + k0 + akb, arow1 + k0 + akb + 16);
    WAIT_ASYNC0();                      // my copies for cur landed
    __syncthreads();                    // everyone's copies landed + prev reads done
    if (k0 + 32 < D_) stageB(k0 + 32, cur ^ 1);  // overlaps compute below
#pragma unroll
    for (int nt = 0; nt < 4; nt++) {
      v16bf bf = load16(&ldsB[cur][(nt * 16 + row16) * 32 + bkb]);
      acc[0][nt] = wmma_bf16(a0, bf, acc[0][nt]);
      acc[1][nt] = wmma_bf16(a1, bf, acc[1][nt]);
    }
    cur ^= 1;
  }

  const int which = nbase >> 10;  // 0=Q 1=K 2=V (uniform: 64 | 1024)
#pragma unroll
  for (int half = 0; half < 2; half++) {
#pragma unroll
    for (int nt = 0; nt < 4; nt++) {
      int n = nbase + nt * 16 + row16;
      float bv = bias[n];
      int dm = n & (D_ - 1), hh = dm >> 6, dd = dm & 63;
#pragma unroll
      for (int j = 0; j < 8; j++) {
        int m  = mbase + half * 16 + j + 8 * hi;
        int bb = m >> 11, ss = m & (S_ - 1);
        float v = acc[half][nt][j] + bv;
        if (which == 0)
          Qb[(((size_t)(bb * H_ + hh) * S_) + ss) * DEPTH_ + dd] = (__bf16)v;
        else if (which == 1)
          Kb[(((size_t)(bb * H_ + hh) * S_) + ss) * DEPTH_ + dd] = (__bf16)v;
        else
          Vt[(((size_t)(bb * H_ + hh) * DEPTH_) + dd) * S_ + ss] = (__bf16)v;
      }
    }
  }
}

// ---------- fused flash attention ----------
// Block = 8 waves, all on the same (b,h); wave w owns q-tile qtbase+w.
// 64-key K/V chunks moved by the Tensor Data Mover: warp 0 issues one descriptor
// per tile (K chunk = contiguous 8KB run; V chunk = 64x128B tile, 4KB row stride),
// waits TENSORcnt, and the workgroup barrier publishes the buffer. Double-buffered.
__global__ __launch_bounds__(256) void k_attention(
    const __bf16* __restrict__ Qb, const __bf16* __restrict__ Kb,
    const __bf16* __restrict__ Vt, const int* __restrict__ mask,
    __bf16* __restrict__ ao) {
  __shared__ __align__(16) __bf16 ldsK[2][64 * 64];   // keys x depth
  __shared__ __align__(16) __bf16 ldsV[2][64 * 64];   // depth x keys
  __shared__ __align__(16) __bf16 pstage[8][16 * 64]; // per-wave P staging
  const int lane = threadIdx.x & 31;
  const int warp = threadIdx.x >> 5;
  const int b  = blockIdx.x >> 8;
  const int h  = (blockIdx.x >> 4) & (H_ - 1);
  const int qt = (blockIdx.x & 15) * 8 + warp;
  const int row16 = lane & 15;
  const int hi    = lane >> 4;
  const int akb   = hi ? 8 : 0;
  const int bkb   = hi ? 8 : 0;
  __bf16* my = &pstage[warp][0];

  const __bf16* kbase = Kb + ((size_t)(b * H_ + h) * S_) * DEPTH_;
  const __bf16* vbase = Vt + ((size_t)(b * H_ + h) * DEPTH_) * S_;
  const int* mrow = mask + b * S_;

  // TDM staging of one 64-key chunk (issued by warp 0 only)
  auto stage = [&](int kc, int bufi) {
    // K rows kc..kc+63 are contiguous: one 4096-element run
    tdm_load_2d((unsigned)(size_t)&ldsK[bufi][0], kbase + (size_t)kc * DEPTH_,
                /*tile_d0=*/64 * 64, /*tile_d1=*/1,
                /*tensor_d0=*/S_ * DEPTH_, /*tensor_d1=*/1, /*stride0=*/S_ * DEPTH_);
    // V: 64 depth-rows x 64 keys, row stride S_ elements
    tdm_load_2d((unsigned)(size_t)&ldsV[bufi][0], vbase + kc,
                /*tile_d0=*/64, /*tile_d1=*/64,
                /*tensor_d0=*/S_, /*tensor_d1=*/DEPTH_, /*stride0=*/S_);
  };

  // Q A-fragments (loop invariant)
  const __bf16* qrow = Qb + (((size_t)(b * H_ + h) * S_) + qt * 16 + row16) * DEPTH_;
  v16bf qa0 = load8x2(qrow + akb,      qrow + akb + 16);
  v16bf qa1 = load8x2(qrow + 32 + akb, qrow + 32 + akb + 16);

  v8f o[4] = {};
  float m8[8], l8[8];
#pragma unroll
  for (int j = 0; j < 8; j++) { m8[j] = -1e30f; l8[j] = 0.f; }

  if (warp == 0) stage(0, 0);
  int cur = 0;
  for (int kc = 0; kc < S_; kc += 64) {
    if (warp == 0) __builtin_amdgcn_s_wait_tensorcnt(0);  // cur tiles landed
    __syncthreads();                    // publish buffer; prev reads done
    if (kc + 64 < S_ && warp == 0) stage(kc + 64, cur ^ 1);  // overlaps compute

    // ---- logits from LDS K: 4 tiles of 16 keys, depth 64 = 2 WMMAs each ----
    v8f s[4];
#pragma unroll
    for (int t = 0; t < 4; t++) {
      const __bf16* krow = &ldsK[cur][(t * 16 + row16) * 64];
      v8f c = {};
      c = wmma_bf16(qa0, load16(krow + bkb), c);
      c = wmma_bf16(qa1, load16(krow + 32 + bkb), c);
      s[t] = c;
    }
    // ---- scale + mask, already in log2 domain ----
#pragma unroll
    for (int t = 0; t < 4; t++) {
      float mv = (float)mrow[kc + t * 16 + row16] * MASK_LOG2E;
#pragma unroll
      for (int j = 0; j < 8; j++) s[t][j] = s[t][j] * SCALE_LOG2E + mv;
    }
    // ---- online softmax per row (16-lane butterfly reductions, exp2) ----
#pragma unroll
    for (int j = 0; j < 8; j++) {
      float rm = fmaxf(fmaxf(s[0][j], s[1][j]), fmaxf(s[2][j], s[3][j]));
#pragma unroll
      for (int msk = 1; msk < 16; msk <<= 1) rm = fmaxf(rm, __shfl_xor(rm, msk, 32));
      float mn    = fmaxf(m8[j], rm);
      float scale = __builtin_amdgcn_exp2f(m8[j] - mn);
      float rs = 0.f;
#pragma unroll
      for (int t = 0; t < 4; t++) {
        float p = __builtin_amdgcn_exp2f(s[t][j] - mn);
        s[t][j] = p;
        rs += p;
      }
#pragma unroll
      for (int msk = 1; msk < 16; msk <<= 1) rs += __shfl_xor(rs, msk, 32);
      l8[j] = l8[j] * scale + rs;
      m8[j] = mn;
#pragma unroll
      for (int nt = 0; nt < 4; nt++) o[nt][j] *= scale;
    }
    // ---- stage P (C-layout -> LDS row-major 16x64, wave-private) ----
#pragma unroll
    for (int t = 0; t < 4; t++)
#pragma unroll
      for (int j = 0; j < 8; j++)
        my[(j + 8 * hi) * 64 + t * 16 + row16] = (__bf16)s[t][j];
    asm volatile("s_wait_dscnt 0x0" ::: "memory");
    const __bf16* prow = my + row16 * 64;
    v16bf pa0 = load8x2(prow + akb,      prow + akb + 16);
    v16bf pa1 = load8x2(prow + 32 + akb, prow + 32 + akb + 16);
    // ---- O += P @ V from LDS V (depth-major => contiguous B-frags) ----
#pragma unroll
    for (int nt = 0; nt < 4; nt++) {
      const __bf16* vcol = &ldsV[cur][(nt * 16 + row16) * 64];
      o[nt] = wmma_bf16(pa0, load16(vcol + bkb), o[nt]);
      o[nt] = wmma_bf16(pa1, load16(vcol + 32 + bkb), o[nt]);
    }
    cur ^= 1;
  }

  // ---- epilogue: O / l -> attn_out bf16 [M, D] (heads re-concatenated) ----
#pragma unroll
  for (int j = 0; j < 8; j++) {
    float inv = 1.0f / l8[j];
    int m = b * S_ + qt * 16 + j + 8 * hi;
#pragma unroll
    for (int nt = 0; nt < 4; nt++) {
      int n = h * DEPTH_ + nt * 16 + row16;
      ao[(size_t)m * D_ + n] = (__bf16)(o[nt][j] * inv);
    }
  }
}

// ---------- output projection: [4096,1024] @ [1024,1024] + bias -> f32 ----------
__global__ __launch_bounds__(256) void k_gemm_out(
    const __bf16* __restrict__ ao, const __bf16* __restrict__ wT,
    const float* __restrict__ bias, float* __restrict__ out) {
  __shared__ __align__(16) __bf16 ldsB[2][64 * 32];
  const int lane  = threadIdx.x & 31;
  const int warp  = threadIdx.x >> 5;
  const int NTG   = D_ / 64;            // 16
  const int mtg   = blockIdx.x / NTG;
  const int ng    = blockIdx.x % NTG;
  const int mbase = (mtg * 8 + warp) * 32;
  const int row16 = lane & 15;
  const int hi    = lane >> 4;
  const int akb   = hi ? 8 : 0;
  const int bkb   = hi ? 8 : 0;
  const int nbase = ng * 64;

  const int sr = threadIdx.x >> 2;
  const int sc = (threadIdx.x & 3) * 8;
  auto stageB = [&](int k0, int bufi) {
    async_b128(wT + (size_t)(nbase + sr) * D_ + k0 + sc, &ldsB[bufi][sr * 32 + sc]);
  };

  v8f acc[2][4] = {};
  const __bf16* arow0 = ao + (size_t)(mbase + row16) * D_;
  const __bf16* arow1 = arow0 + (size_t)16 * D_;

  stageB(0, 0);
  int cur = 0;
  for (int k0 = 0; k0 < D_; k0 += 32) {
    v16bf a0 = load8x2(arow0 + k0 + akb, arow0 + k0 + akb + 16);
    v16bf a1 = load8x2(arow1 + k0 + akb, arow1 + k0 + akb + 16);
    WAIT_ASYNC0();
    __syncthreads();
    if (k0 + 32 < D_) stageB(k0 + 32, cur ^ 1);
#pragma unroll
    for (int nt = 0; nt < 4; nt++) {
      v16bf bf = load16(&ldsB[cur][(nt * 16 + row16) * 32 + bkb]);
      acc[0][nt] = wmma_bf16(a0, bf, acc[0][nt]);
      acc[1][nt] = wmma_bf16(a1, bf, acc[1][nt]);
    }
    cur ^= 1;
  }
#pragma unroll
  for (int half = 0; half < 2; half++) {
#pragma unroll
    for (int nt = 0; nt < 4; nt++) {
      int n = nbase + nt * 16 + row16;
      float bv = bias[n];
#pragma unroll
      for (int j = 0; j < 8; j++) {
        int m = mbase + half * 16 + j + 8 * hi;
        out[(size_t)m * D_ + n] = acc[half][nt][j] + bv;
      }
    }
  }
}

// ---------- launch ----------
extern "C" void kernel_launch(void* const* d_in, const int* in_sizes, int n_in,
                              void* d_out, int out_size, void* d_ws, size_t ws_size,
                              hipStream_t stream) {
  const float* x     = (const float*)d_in[0];
  const int*   mask  = (const int*)d_in[1];
  const float* w_qkv = (const float*)d_in[2];
  const float* b_qkv = (const float*)d_in[3];
  const float* w_out = (const float*)d_in[4];
  const float* b_out = (const float*)d_in[5];
  float* out = (float*)d_out;

  char* p = (char*)d_ws;
  __bf16* xb    = (__bf16*)p; p += (size_t)M_ * D_ * 2;                 // 8 MB
  __bf16* wqkvT = (__bf16*)p; p += (size_t)N3_ * D_ * 2;                // 6 MB
  __bf16* wOutT = (__bf16*)p; p += (size_t)D_ * D_ * 2;                 // 2 MB
  __bf16* Qb    = (__bf16*)p; p += (size_t)B_ * H_ * S_ * DEPTH_ * 2;   // 8 MB
  __bf16* Kb    = (__bf16*)p; p += (size_t)B_ * H_ * S_ * DEPTH_ * 2;   // 8 MB
  __bf16* Vt    = (__bf16*)p; p += (size_t)B_ * H_ * S_ * DEPTH_ * 2;   // 8 MB
  __bf16* ao    = (__bf16*)p; p += (size_t)M_ * D_ * 2;                 // 8 MB

  dim3 blk(256);
  k_cvt_bf16<<<(M_ * D_ + 255) / 256, blk, 0, stream>>>(x, xb, M_ * D_);
  k_transpose_bf16<<<(D_ * N3_ + 255) / 256, blk, 0, stream>>>(w_qkv, wqkvT, D_, N3_);
  k_transpose_bf16<<<(D_ * D_ + 255) / 256, blk, 0, stream>>>(w_out, wOutT, D_, D_);

  // (4096/256) m-groups * 48 n-groups = 768 blocks
  k_gemm_qkv<<<16 * 48, blk, 0, stream>>>(xb, wqkvT, b_qkv, Qb, Kb, Vt);
  // B*H*(S/128) = 512 blocks, 8 q-tiles each
  k_attention<<<B_ * H_ * (S_ / 128), blk, 0, stream>>>(Qb, Kb, Vt, mask, ao);
  // (4096/256) * 16 = 256 blocks
  k_gemm_out<<<16 * 16, blk, 0, stream>>>(ao, wOutT, b_out, out);
}
